// Flow_Attention_71201967833539
// MI455X (gfx1250) — matrix operand order repaired and checked by reading, not compiled, not run on hardware
//
#include <hip/hip_runtime.h>
#include <hip/hip_bf16.h>

// ---------------------------------------------------------------------------
// Flow-Attention for MI455X (gfx1250, wave32, WMMA).
// B=4, N=4096, C=768, H=12, d=64.  All matmul-shaped work goes through
// v_wmma_f32_16x16x32_bf16 with double-buffered fragment loads; irregular
// N-reductions use one block per (b,h) with wave shuffles + LDS atomics.
// ---------------------------------------------------------------------------

#define EPSF 1e-6f

typedef __attribute__((ext_vector_type(16))) __bf16 v16bf;
typedef __attribute__((ext_vector_type(8)))  float  v8f;

union FragB16 { v16bf v; uint4 q[2]; };

__device__ __forceinline__ unsigned short f2bf(float f) {
  unsigned int x = __float_as_uint(f);
  x += 0x7FFFu + ((x >> 16) & 1u);           // round-to-nearest-even
  return (unsigned short)(x >> 16);
}

__device__ __forceinline__ float sigm(float x) { return 1.0f / (1.0f + __expf(-x)); }

__device__ __forceinline__ float waveSum(float x) {
  x += __shfl_xor(x, 16, 32);
  x += __shfl_xor(x, 8, 32);
  x += __shfl_xor(x, 4, 32);
  x += __shfl_xor(x, 2, 32);
  x += __shfl_xor(x, 1, 32);
  return x;
}

// Load one 2-A + 4-B fragment set at k-offset kk.
__device__ __forceinline__ void g_load6(
    FragB16& a0, FragB16& a1, FragB16& b0, FragB16& b1, FragB16& b2, FragB16& b3,
    const unsigned short* arow0, const unsigned short* arow1,
    const unsigned short* brow0, const unsigned short* brow1,
    const unsigned short* brow2, const unsigned short* brow3, int kk, int hi) {
  a0.q[0] = *(const uint4*)(arow0 + kk + hi * 16);
  a0.q[1] = *(const uint4*)(arow0 + kk + 8 + hi * 16);
  a1.q[0] = *(const uint4*)(arow1 + kk + hi * 16);
  a1.q[1] = *(const uint4*)(arow1 + kk + 8 + hi * 16);
  b0.q[0] = *(const uint4*)(brow0 + kk); b0.q[1] = *(const uint4*)(brow0 + kk + 8);
  b1.q[0] = *(const uint4*)(brow1 + kk); b1.q[1] = *(const uint4*)(brow1 + kk + 8);
  b2.q[0] = *(const uint4*)(brow2 + kk); b2.q[1] = *(const uint4*)(brow2 + kk + 8);
  b3.q[0] = *(const uint4*)(brow3 + kk); b3.q[1] = *(const uint4*)(brow3 + kk + 8);
}

__device__ __forceinline__ void g_mma8(
    v8f acc[2][4], const FragB16& a0, const FragB16& a1,
    const FragB16& b0, const FragB16& b1, const FragB16& b2, const FragB16& b3) {
  acc[0][0] = __builtin_amdgcn_wmma_f32_16x16x32_bf16(false, a0.v, false, b0.v, (short)0, acc[0][0], false, false);
  acc[1][0] = __builtin_amdgcn_wmma_f32_16x16x32_bf16(false, a1.v, false, b0.v, (short)0, acc[1][0], false, false);
  acc[0][1] = __builtin_amdgcn_wmma_f32_16x16x32_bf16(false, a0.v, false, b1.v, (short)0, acc[0][1], false, false);
  acc[1][1] = __builtin_amdgcn_wmma_f32_16x16x32_bf16(false, a1.v, false, b1.v, (short)0, acc[1][1], false, false);
  acc[0][2] = __builtin_amdgcn_wmma_f32_16x16x32_bf16(false, a0.v, false, b2.v, (short)0, acc[0][2], false, false);
  acc[1][2] = __builtin_amdgcn_wmma_f32_16x16x32_bf16(false, a1.v, false, b2.v, (short)0, acc[1][2], false, false);
  acc[0][3] = __builtin_amdgcn_wmma_f32_16x16x32_bf16(false, a0.v, false, b3.v, (short)0, acc[0][3], false, false);
  acc[1][3] = __builtin_amdgcn_wmma_f32_16x16x32_bf16(false, a1.v, false, b3.v, (short)0, acc[1][3], false, false);
}

// Single-A variant (16 rows/wave) for the per-head kv_agg GEMM.
__device__ __forceinline__ void g_load5(
    FragB16& a0, FragB16& b0, FragB16& b1, FragB16& b2, FragB16& b3,
    const unsigned short* arow0,
    const unsigned short* brow0, const unsigned short* brow1,
    const unsigned short* brow2, const unsigned short* brow3, int kk, int hi) {
  a0.q[0] = *(const uint4*)(arow0 + kk + hi * 16);
  a0.q[1] = *(const uint4*)(arow0 + kk + 8 + hi * 16);
  b0.q[0] = *(const uint4*)(brow0 + kk); b0.q[1] = *(const uint4*)(brow0 + kk + 8);
  b1.q[0] = *(const uint4*)(brow1 + kk); b1.q[1] = *(const uint4*)(brow1 + kk + 8);
  b2.q[0] = *(const uint4*)(brow2 + kk); b2.q[1] = *(const uint4*)(brow2 + kk + 8);
  b3.q[0] = *(const uint4*)(brow3 + kk); b3.q[1] = *(const uint4*)(brow3 + kk + 8);
}

__device__ __forceinline__ void g_mma4(
    v8f acc[4], const FragB16& a0,
    const FragB16& b0, const FragB16& b1, const FragB16& b2, const FragB16& b3) {
  acc[0] = __builtin_amdgcn_wmma_f32_16x16x32_bf16(false, a0.v, false, b0.v, (short)0, acc[0], false, false);
  acc[1] = __builtin_amdgcn_wmma_f32_16x16x32_bf16(false, a0.v, false, b1.v, (short)0, acc[1], false, false);
  acc[2] = __builtin_amdgcn_wmma_f32_16x16x32_bf16(false, a0.v, false, b2.v, (short)0, acc[2], false, false);
  acc[3] = __builtin_amdgcn_wmma_f32_16x16x32_bf16(false, a0.v, false, b3.v, (short)0, acc[3], false, false);
}

// ---------------------------------------------------------------------------
// fp32 -> bf16 conversion
// ---------------------------------------------------------------------------
__global__ __launch_bounds__(256) void cvt_bf16_kernel(const float* __restrict__ in,
                                                       unsigned short* __restrict__ out,
                                                       int n) {
  int i = blockIdx.x * 256 + threadIdx.x;
  if (i < n) out[i] = f2bf(in[i]);
}

// ---------------------------------------------------------------------------
// 768-K WMMA GEMM:  Y[m, o] = sigmoid?( sum_c X[m,c]*W[o,c] + bias[o] )
// Block: 128 threads (4 waves). Tile: 128(M) x 64(N). Each wave: 32 rows x
// 64 cols (8 accumulators), K step 32, ping-pong double-buffered fragments.
// ---------------------------------------------------------------------------
__global__ __launch_bounds__(128) void wmma_gemm768_kernel(
    const unsigned short* __restrict__ X,   // [M,768] bf16
    const unsigned short* __restrict__ W,   // [768,768] bf16, row-major [o][c]
    const float* __restrict__ bias,         // [768]
    float* __restrict__ Y,
    int applySigmoid, int headLayout) {
  const int K    = 768;
  const int lane = threadIdx.x & 31;
  const int wv   = threadIdx.x >> 5;
  const int lo   = lane & 15;
  const int hi   = lane >> 4;
  const int rowBase = blockIdx.x * 128 + wv * 32;
  const int colBase = blockIdx.y * 64;

  v8f acc[2][4];
#pragma unroll
  for (int t = 0; t < 2; ++t)
#pragma unroll
    for (int j = 0; j < 4; ++j)
#pragma unroll
      for (int r = 0; r < 8; ++r) acc[t][j][r] = 0.0f;

  const unsigned short* arow0 = X + (size_t)(rowBase + lo) * K;
  const unsigned short* arow1 = X + (size_t)(rowBase + 16 + lo) * K;
  const unsigned short* brow0 = W + (size_t)(colBase + 0 * 16 + lo) * K + hi * 16;
  const unsigned short* brow1 = W + (size_t)(colBase + 1 * 16 + lo) * K + hi * 16;
  const unsigned short* brow2 = W + (size_t)(colBase + 2 * 16 + lo) * K + hi * 16;
  const unsigned short* brow3 = W + (size_t)(colBase + 3 * 16 + lo) * K + hi * 16;

  FragB16 pa0, pa1, pb0, pb1, pb2, pb3;   // ping
  FragB16 qa0, qa1, qb0, qb1, qb2, qb3;   // pong
  g_load6(pa0, pa1, pb0, pb1, pb2, pb3, arow0, arow1, brow0, brow1, brow2, brow3, 0, hi);

  for (int kk = 0; kk < K; kk += 64) {     // 24 k-steps -> 12 double-steps
    __builtin_prefetch(arow0 + kk + 512, 0, 1);   // global_prefetch_b8
    g_load6(qa0, qa1, qb0, qb1, qb2, qb3, arow0, arow1, brow0, brow1, brow2, brow3,
            kk + 32, hi);
    g_mma8(acc, pa0, pa1, pb0, pb1, pb2, pb3);
    if (kk + 64 < K) {
      g_load6(pa0, pa1, pb0, pb1, pb2, pb3, arow0, arow1, brow0, brow1, brow2, brow3,
              kk + 64, hi);
    }
    g_mma8(acc, qa0, qa1, qb0, qb1, qb2, qb3);
  }

#pragma unroll
  for (int t = 0; t < 2; ++t) {
#pragma unroll
    for (int j = 0; j < 4; ++j) {
      const int col = colBase + j * 16 + lo;
      const float bb = bias[col];
#pragma unroll
      for (int r = 0; r < 8; ++r) {
        const int m = rowBase + t * 16 + r + 8 * hi;
        float v = acc[t][j][r] + bb;
        if (applySigmoid) v = sigm(v);
        size_t idx;
        if (headLayout) {
          const int b = m >> 12, n = m & 4095;
          const int h = col >> 6, dd = col & 63;
          idx = ((((size_t)b * 12 + h) * 4096) + n) * 64 + dd;
        } else {
          idx = (size_t)m * 768 + col;
        }
        Y[idx] = v;
      }
    }
  }
}

// ---------------------------------------------------------------------------
// Flow-stats: one block (512 thr = 16 waves) per (b,h). Lane owns d0=lane,
// d1=lane+32; wave g handles rows n = g, g+16, ... Wave shuffle for d-dots,
// LDS atomics for cross-wave [d]-vector reductions.
// Updates qh/kh in place to qh'/kh'; writes scale_q (=sink_in*alloc) and
// comp (=softmax(conserved_source)*N).
// ---------------------------------------------------------------------------
__global__ __launch_bounds__(512) void flow_stats_kernel(
    float* __restrict__ qh, float* __restrict__ kh, const float* __restrict__ mh,
    float* __restrict__ sink_in, float* __restrict__ scale_q, float* __restrict__ comp) {
  __shared__ float s_msum[64], s_tq[64], s_tk[64], s_qsum[64], s_ksum[64], s_uq[64], s_uk[64];
  __shared__ float s_sum;
  const int tid  = threadIdx.x;
  const int lane = tid & 31;
  const int g    = tid >> 5;
  const int d0   = lane, d1 = lane + 32;
  const int bh   = blockIdx.x;
  const size_t base = (size_t)bh * 4096 * 64;
  float* Q        = qh + base;
  float* Kp       = kh + base;
  const float* Mm = mh + base;
  float* SI = sink_in + (size_t)bh * 4096;
  float* SQ = scale_q + (size_t)bh * 4096;
  float* CP = comp + (size_t)bh * 4096;

  if (tid < 64) {
    s_msum[tid] = 0.f; s_tq[tid] = 0.f; s_tk[tid] = 0.f;
    s_qsum[tid] = 0.f; s_ksum[tid] = 0.f; s_uq[tid] = 0.f; s_uk[tid] = 0.f;
  }
  if (tid == 0) s_sum = 0.f;
  __syncthreads();

  // Phase 0: m_sum = sum_n mh
  {
    float a0 = 0.f, a1 = 0.f;
    for (int n = g; n < 4096; n += 16) { a0 += Mm[n * 64 + d0]; a1 += Mm[n * 64 + d1]; }
    atomicAdd(&s_msum[d0], a0); atomicAdd(&s_msum[d1], a1);
  }
  __syncthreads();

  // Phase 1: eq/ek per row, accumulate t_q = sum qh*eq, t_k = sum kh*ek
  {
    const float ms0 = s_msum[d0] + EPSF, ms1 = s_msum[d1] + EPSF;
    float tq0 = 0, tq1 = 0, tk0 = 0, tk1 = 0;
    for (int n = g; n < 4096; n += 16) {
      const float q0 = Q[n * 64 + d0], q1 = Q[n * 64 + d1];
      const float eq = 1.0f / (waveSum((q0 + EPSF) * ms0 + (q1 + EPSF) * ms1) + EPSF);
      tq0 += q0 * eq; tq1 += q1 * eq;
      const float k0 = Kp[n * 64 + d0], k1 = Kp[n * 64 + d1];
      const float ek = 1.0f / (waveSum((k0 + EPSF) * ms0 + (k1 + EPSF) * ms1) + EPSF);
      tk0 += k0 * ek; tk1 += k1 * ek;
    }
    atomicAdd(&s_tq[d0], tq0); atomicAdd(&s_tq[d1], tq1);
    atomicAdd(&s_tk[d0], tk0); atomicAdd(&s_tk[d1], tk1);
  }
  __syncthreads();

  // Phase 2: eqq/ekk, qh' = qh*(1+mh*eqq), kh' = kh*(1+mh*ekk); sums over n
  {
    const float tq0e = s_tq[d0] + EPSF, tq1e = s_tq[d1] + EPSF;
    const float tk0e = s_tk[d0] + EPSF, tk1e = s_tk[d1] + EPSF;
    float qs0 = 0, qs1 = 0, ks0 = 0, ks1 = 0;
    for (int n = g; n < 4096; n += 16) {
      const float m0 = Mm[n * 64 + d0], m1 = Mm[n * 64 + d1];
      const float eqq = sigm(waveSum((m0 + EPSF) * tq0e + (m1 + EPSF) * tq1e) + EPSF);
      const float ekk = sigm(waveSum((m0 + EPSF) * tk0e + (m1 + EPSF) * tk1e) + EPSF);
      const float q0 = Q[n * 64 + d0] * (1.0f + m0 * eqq);
      const float q1 = Q[n * 64 + d1] * (1.0f + m1 * eqq);
      Q[n * 64 + d0] = q0; Q[n * 64 + d1] = q1;
      qs0 += q0; qs1 += q1;
      const float k0 = Kp[n * 64 + d0] * (1.0f + m0 * ekk);
      const float k1 = Kp[n * 64 + d1] * (1.0f + m1 * ekk);
      Kp[n * 64 + d0] = k0; Kp[n * 64 + d1] = k1;
      ks0 += k0; ks1 += k1;
    }
    atomicAdd(&s_qsum[d0], qs0); atomicAdd(&s_qsum[d1], qs1);
    atomicAdd(&s_ksum[d0], ks0); atomicAdd(&s_ksum[d1], ks1);
  }
  __syncthreads();

  // Phase 3: sink_incoming / source_outgoing; accumulate u_q, u_k
  {
    const float kse0 = s_ksum[d0] + EPSF, kse1 = s_ksum[d1] + EPSF;
    const float qse0 = s_qsum[d0] + EPSF, qse1 = s_qsum[d1] + EPSF;
    float uq0 = 0, uq1 = 0, uk0 = 0, uk1 = 0;
    for (int n = g; n < 4096; n += 16) {
      const float q0 = Q[n * 64 + d0], q1 = Q[n * 64 + d1];
      const float si = 1.0f / (waveSum((q0 + EPSF) * kse0 + (q1 + EPSF) * kse1) + EPSF);
      const float k0 = Kp[n * 64 + d0], k1 = Kp[n * 64 + d1];
      const float so = 1.0f / (waveSum((k0 + EPSF) * qse0 + (k1 + EPSF) * qse1) + EPSF);
      if (lane == 0) SI[n] = si;
      uq0 += q0 * si; uq1 += q1 * si;
      uk0 += k0 * so; uk1 += k1 * so;
    }
    atomicAdd(&s_uq[d0], uq0); atomicAdd(&s_uq[d1], uq1);
    atomicAdd(&s_uk[d0], uk0); atomicAdd(&s_uk[d1], uk1);
  }
  __syncthreads();

  // Phase 4: conserved_sink -> sink_allocation; conserved_source (clipped to
  // [-1,1] so softmax needs no max pass) -> exp + sum
  {
    const float uke0 = s_uk[d0] + EPSF, uke1 = s_uk[d1] + EPSF;
    const float uqe0 = s_uq[d0] + EPSF, uqe1 = s_uq[d1] + EPSF;
    float esum = 0.f;
    for (int n = g; n < 4096; n += 16) {
      const float q0 = Q[n * 64 + d0], q1 = Q[n * 64 + d1];
      const float csink = waveSum((q0 + EPSF) * uke0 + (q1 + EPSF) * uke1) + EPSF;
      const float alloc = sigm(csink);   // ratio == 1.0 (N_q == N_k)
      const float k0 = Kp[n * 64 + d0], k1 = Kp[n * 64 + d1];
      float csrc = waveSum((k0 + EPSF) * uqe0 + (k1 + EPSF) * uqe1) + EPSF;
      csrc = fminf(1.0f, fmaxf(-1.0f, csrc));
      const float e = __expf(csrc);
      if (lane == 0) { SQ[n] = SI[n] * alloc; CP[n] = e; esum += e; }
    }
    if (lane == 0) atomicAdd(&s_sum, esum);
  }
  __threadfence_block();
  __syncthreads();

  // Phase 5: comp = exp/sum * N
  {
    const float scale = 4096.0f / s_sum;
    for (int n = tid; n < 4096; n += 512) CP[n] *= scale;
  }
}

// ---------------------------------------------------------------------------
// Prep: bf16 copies for the attention-core GEMMs (grid 48 x 8 slices).
//   qhb[n][d]   = bf16(qh')                     (row-major)
//   khT[d][n]   = bf16(kh')                     (transposed)
//   vwT[d][n]   = bf16(vh * comp[n])            (transposed)
// ---------------------------------------------------------------------------
__global__ __launch_bounds__(256) void prep_kernel(
    const float* __restrict__ qh, const float* __restrict__ kh,
    const float* __restrict__ vh, const float* __restrict__ comp,
    unsigned short* __restrict__ qhb, unsigned short* __restrict__ khT,
    unsigned short* __restrict__ vwT) {
  const int bh = blockIdx.x;
  const size_t base = (size_t)bh * 4096 * 64;
  const float* Q  = qh + base;
  const float* Kp = kh + base;
  const float* V  = vh + base;
  const float* CP = comp + (size_t)bh * 4096;
  unsigned short* qb = qhb + base;
  unsigned short* kT = khT + base;   // [64][4096] per bh
  unsigned short* vT = vwT + base;   // [64][4096] per bh
  const int slice = (4096 * 64) / 8;               // 32768
  const int start = blockIdx.y * slice;
  for (int idx = start + threadIdx.x; idx < start + slice; idx += 256) {
    const int n = idx >> 6, d = idx & 63;
    const float c = CP[n];
    qb[idx] = f2bf(Q[idx]);
    kT[(size_t)d * 4096 + n] = f2bf(Kp[idx]);
    vT[(size_t)d * 4096 + n] = f2bf(V[idx] * c);
  }
}

// ---------------------------------------------------------------------------
// kv_agg partials:  part[kz][bh][e*64+d] = sum_{n in chunk kz} kh'[n,d]*vw[n,e]
// (per head 64x64 over a 512-wide K chunk; grid 48 x 8, 4 waves/block,
//  double-buffered fragments)
// ---------------------------------------------------------------------------
__global__ __launch_bounds__(128) void kvagg_part_kernel(
    const unsigned short* __restrict__ khT, const unsigned short* __restrict__ vwT,
    float* __restrict__ part) {
  const int lane = threadIdx.x & 31;
  const int wv   = threadIdx.x >> 5;
  const int lo   = lane & 15, hi = lane >> 4;
  const int bh   = blockIdx.x;
  const int kz   = blockIdx.y;
  const int k0   = kz * 512;

  v8f acc[4];
#pragma unroll
  for (int j = 0; j < 4; ++j)
#pragma unroll
    for (int r = 0; r < 8; ++r) acc[j][r] = 0.0f;

  const unsigned short* arow0 = khT + ((size_t)bh * 64 + wv * 16 + lo) * 4096 + k0;
  const unsigned short* brow0 = vwT + ((size_t)bh * 64 + 0 * 16 + lo) * 4096 + k0 + hi * 16;
  const unsigned short* brow1 = vwT + ((size_t)bh * 64 + 1 * 16 + lo) * 4096 + k0 + hi * 16;
  const unsigned short* brow2 = vwT + ((size_t)bh * 64 + 2 * 16 + lo) * 4096 + k0 + hi * 16;
  const unsigned short* brow3 = vwT + ((size_t)bh * 64 + 3 * 16 + lo) * 4096 + k0 + hi * 16;

  FragB16 pa, pb0, pb1, pb2, pb3;
  FragB16 qa, qb0, qb1, qb2, qb3;
  g_load5(pa, pb0, pb1, pb2, pb3, arow0, brow0, brow1, brow2, brow3, 0, hi);

  for (int kk = 0; kk < 512; kk += 64) {   // 16 k-steps -> 8 double-steps
    g_load5(qa, qb0, qb1, qb2, qb3, arow0, brow0, brow1, brow2, brow3, kk + 32, hi);
    g_mma4(acc, pa, pb0, pb1, pb2, pb3);
    if (kk + 64 < 512) {
      g_load5(pa, pb0, pb1, pb2, pb3, arow0, brow0, brow1, brow2, brow3, kk + 64, hi);
    }
    g_mma4(acc, qa, qb0, qb1, qb2, qb3);
  }

  float* dst = part + ((size_t)kz * 48 + bh) * 4096;
#pragma unroll
  for (int j = 0; j < 4; ++j) {
    const int e = j * 16 + lo;
#pragma unroll
    for (int r = 0; r < 8; ++r) {
      const int d = wv * 16 + r + 8 * hi;
      dst[e * 64 + d] = acc[j][r];       // stored transposed: [e][d]
    }
  }
}

// Deterministic reduction of the 8 K-chunk partials -> bf16 kvT[bh][e][d].
__global__ __launch_bounds__(256) void kvagg_reduce_kernel(
    const float* __restrict__ part, unsigned short* __restrict__ kvT) {
  const int bh = blockIdx.x;
  for (int idx = threadIdx.x; idx < 4096; idx += 256) {
    float s = 0.f;
#pragma unroll
    for (int kz = 0; kz < 8; ++kz) s += part[((size_t)kz * 48 + bh) * 4096 + idx];
    kvT[(size_t)bh * 4096 + idx] = f2bf(s);
  }
}

// ---------------------------------------------------------------------------
// x[n,e] = (sum_d qh'[n,d] * kv_agg[d,e]) * scale_q[n]   (per head: 4096x64, K=64)
// Stored into xb[b,n, h*64+e] (bf16) for the final output projection.
// ---------------------------------------------------------------------------
__global__ __launch_bounds__(128) void xcalc_kernel(
    const unsigned short* __restrict__ qhb, const unsigned short* __restrict__ kvT,
    const float* __restrict__ scale_q, unsigned short* __restrict__ xb) {
  const int lane = threadIdx.x & 31;
  const int wv   = threadIdx.x >> 5;
  const int lo   = lane & 15, hi = lane >> 4;
  const int bh   = blockIdx.y;
  const int b    = bh / 12, h = bh % 12;
  const int rowBase = blockIdx.x * 64 + wv * 16;

  v8f acc[4];
#pragma unroll
  for (int j = 0; j < 4; ++j)
#pragma unroll
    for (int r = 0; r < 8; ++r) acc[j][r] = 0.0f;

  // Preload both K=64 fragment sets (2 k-steps), then issue all 8 WMMAs.
  const unsigned short* arow = qhb + ((size_t)bh * 4096 + rowBase + lo) * 64;
  FragB16 a0, a1;
  a0.q[0] = *(const uint4*)(arow + hi * 16);
  a0.q[1] = *(const uint4*)(arow + 8 + hi * 16);
  a1.q[0] = *(const uint4*)(arow + 32 + hi * 16);
  a1.q[1] = *(const uint4*)(arow + 40 + hi * 16);
#pragma unroll
  for (int j = 0; j < 4; ++j) {
    const unsigned short* brow = kvT + ((size_t)bh * 64 + j * 16 + lo) * 64 + hi * 16;
    FragB16 b0, b1;
    b0.q[0] = *(const uint4*)(brow);      b0.q[1] = *(const uint4*)(brow + 8);
    b1.q[0] = *(const uint4*)(brow + 32); b1.q[1] = *(const uint4*)(brow + 40);
    acc[j] = __builtin_amdgcn_wmma_f32_16x16x32_bf16(false, a0.v, false, b0.v, (short)0, acc[j], false, false);
    acc[j] = __builtin_amdgcn_wmma_f32_16x16x32_bf16(false, a1.v, false, b1.v, (short)0, acc[j], false, false);
  }
#pragma unroll
  for (int j = 0; j < 4; ++j) {
    const int e = j * 16 + lo;
#pragma unroll
    for (int r = 0; r < 8; ++r) {
      const int n = rowBase + r + 8 * hi;
      const float s = scale_q[(size_t)bh * 4096 + n];
      xb[((size_t)(b * 4096 + n)) * 768 + h * 64 + e] = f2bf(acc[j][r] * s);
    }
  }
}

// ---------------------------------------------------------------------------
// Host-side launcher
// ---------------------------------------------------------------------------
extern "C" void kernel_launch(void* const* d_in, const int* in_sizes, int n_in,
                              void* d_out, int out_size, void* d_ws, size_t ws_size,
                              hipStream_t stream) {
  (void)in_sizes; (void)n_in; (void)out_size; (void)ws_size;

  const float* q_in  = (const float*)d_in[0];
  const float* kv_in = (const float*)d_in[1];
  const float* mo_in = (const float*)d_in[2];
  const float* Wq = (const float*)d_in[3];  const float* bq = (const float*)d_in[4];
  const float* Wk = (const float*)d_in[5];  const float* bk = (const float*)d_in[6];
  const float* Wv = (const float*)d_in[7];  const float* bv = (const float*)d_in[8];
  const float* Wm = (const float*)d_in[9];  const float* bm = (const float*)d_in[10];
  const float* Wo = (const float*)d_in[11]; const float* bo = (const float*)d_in[12];

  char* ws = (char*)d_ws;
  constexpr size_t SZ_HEAD = (size_t)48 * 4096 * 64 * 4;   // 50,331,648 B (fp32 heads)
  constexpr size_t SZ_XBF  = (size_t)16384 * 768 * 2;      // 25,165,824 B (bf16 [M,768])
  constexpr size_t SZ_KVT  = (size_t)48 * 64 * 64 * 2;     // 393,216 B
  constexpr size_t SZ_WBF  = (size_t)768 * 768 * 2;        // 1,179,648 B
  constexpr size_t SZ_SCAL = (size_t)48 * 4096 * 4;        // 786,432 B

  // Region A..D: fp32 heads
  float* qh = (float*)(ws + 0 * SZ_HEAD);
  float* kh = (float*)(ws + 1 * SZ_HEAD);
  float* vh = (float*)(ws + 2 * SZ_HEAD);
  float* mh = (float*)(ws + 3 * SZ_HEAD);
  // Region D reuse (mh dead after flow_stats): xb + kvT + kv partials
  unsigned short* xb   = (unsigned short*)(ws + 3 * SZ_HEAD);
  unsigned short* kvT  = (unsigned short*)(ws + 3 * SZ_HEAD + SZ_XBF);
  float*          kvP  = (float*)(ws + 3 * SZ_HEAD + SZ_XBF + SZ_KVT);  // 8*48*4096*4 = 6.3 MB
  // Region E: bf16 inputs; reused after projections for qhb/khT/vwT
  constexpr size_t OFF_E = 4 * SZ_HEAD;
  unsigned short* Xq  = (unsigned short*)(ws + OFF_E);
  unsigned short* Xkv = (unsigned short*)(ws + OFF_E + SZ_XBF);
  unsigned short* Xm  = (unsigned short*)(ws + OFF_E + 2 * SZ_XBF);
  unsigned short* qhb = Xq;     // aliases: safe, Xq/Xkv/Xm dead after proj GEMMs
  unsigned short* khT = Xkv;
  unsigned short* vwT = Xm;
  // Region F: bf16 weights
  constexpr size_t OFF_F = OFF_E + 3 * SZ_XBF;
  unsigned short* WqB = (unsigned short*)(ws + OFF_F + 0 * SZ_WBF);
  unsigned short* WkB = (unsigned short*)(ws + OFF_F + 1 * SZ_WBF);
  unsigned short* WvB = (unsigned short*)(ws + OFF_F + 2 * SZ_WBF);
  unsigned short* WmB = (unsigned short*)(ws + OFF_F + 3 * SZ_WBF);
  unsigned short* WoB = (unsigned short*)(ws + OFF_F + 4 * SZ_WBF);
  // Region G: per-row scalars
  constexpr size_t OFF_G = OFF_F + 5 * SZ_WBF;
  float* sink_in = (float*)(ws + OFF_G + 0 * SZ_SCAL);
  float* scale_q = (float*)(ws + OFF_G + 1 * SZ_SCAL);
  float* comp    = (float*)(ws + OFF_G + 2 * SZ_SCAL);

  // 1) fp32 -> bf16 conversions
  const int NIN = 16384 * 768;   // 12,582,912
  const int NW  = 768 * 768;     // 589,824
  cvt_bf16_kernel<<<(NIN + 255) / 256, 256, 0, stream>>>(q_in,  Xq,  NIN);
  cvt_bf16_kernel<<<(NIN + 255) / 256, 256, 0, stream>>>(kv_in, Xkv, NIN);
  cvt_bf16_kernel<<<(NIN + 255) / 256, 256, 0, stream>>>(mo_in, Xm,  NIN);
  cvt_bf16_kernel<<<(NW + 255) / 256, 256, 0, stream>>>(Wq, WqB, NW);
  cvt_bf16_kernel<<<(NW + 255) / 256, 256, 0, stream>>>(Wk, WkB, NW);
  cvt_bf16_kernel<<<(NW + 255) / 256, 256, 0, stream>>>(Wv, WvB, NW);
  cvt_bf16_kernel<<<(NW + 255) / 256, 256, 0, stream>>>(Wm, WmB, NW);
  cvt_bf16_kernel<<<(NW + 255) / 256, 256, 0, stream>>>(Wo, WoB, NW);

  // 2) projections (WMMA): 16384 x 768 x 768 each, head-layout outputs
  dim3 gProj(128, 12);
  wmma_gemm768_kernel<<<gProj, 128, 0, stream>>>(Xq,  WqB, bq, qh, 1, 1);
  wmma_gemm768_kernel<<<gProj, 128, 0, stream>>>(Xkv, WkB, bk, kh, 1, 1);
  wmma_gemm768_kernel<<<gProj, 128, 0, stream>>>(Xkv, WvB, bv, vh, 0, 1);
  wmma_gemm768_kernel<<<gProj, 128, 0, stream>>>(Xm,  WmB, bm, mh, 1, 1);

  // 3) flow statistics + in-place qh'/kh'
  flow_stats_kernel<<<48, 512, 0, stream>>>(qh, kh, mh, sink_in, scale_q, comp);

  // 4) bf16 prep (row-major qh', transposed kh' and vh*comp)
  dim3 gPrep(48, 8);
  prep_kernel<<<gPrep, 256, 0, stream>>>(qh, kh, vh, comp, qhb, khT, vwT);

  // 5) kv_agg (WMMA, per head 64x64, K split into 8 chunks) + deterministic reduce
  dim3 gKv(48, 8);
  kvagg_part_kernel<<<gKv, 128, 0, stream>>>(khT, vwT, kvP);
  kvagg_reduce_kernel<<<48, 256, 0, stream>>>(kvP, kvT);

  // 6) x = qh' @ kv_agg * scale (WMMA, per head 4096x64 K=64) -> [B,N,C] bf16
  dim3 gX(64, 48);
  xcalc_kernel<<<gX, 128, 0, stream>>>(qhb, kvT, scale_q, xb);

  // 7) output projection (WMMA): 16384 x 768 x 768, fp32 row-major to d_out
  wmma_gemm768_kernel<<<gProj, 128, 0, stream>>>(xb, WoB, bo, (float*)d_out, 0, 0);
}
